// ChannelCrossAttention_78116865180380
// MI455X (gfx1250) — compile-verified, matrix operand order checked by reading, not run on hardware
//
#include <hip/hip_runtime.h>

typedef _Float16 v16h __attribute__((ext_vector_type(16)));
typedef float    v8f  __attribute__((ext_vector_type(8)));

#define BATCH 16
#define CH    512      // C1 == C2 == CO
#define NSP   4096     // H*W

union HFrag { uint4 u[2]; v16h v; };      // 16 x f16 = 32B = 8 VGPRs
union HPack8 { uint4 u; _Float16 h[8]; }; // 8 x f16 = 16B

#define WMMA(a, b, c) __builtin_amdgcn_wmma_f32_16x16x32_f16( \
    false, (a), false, (b), (short)0, (c), false, false)

// -----------------------------------------------------------------------------
// K1: fused QKV projection.  C[512 x 4096] = W[512x512] * X[512x4096] + bias
// grid = (32 nTiles, 4 mTiles, 48 = batch*3), block = 256 (8 waves)
// Block tile 128x128; wave tile 32x64 (2x4 WMMA tiles).
// Double-buffered LDS staging (f32 -> f16 convert on stage-in).
// q,k stored row-major [c][n] f16 (via LDS-transposed coalesced b128 stores);
// v stored transposed Vt[n][d] f16 (naturally coalesced from D-layout).
// -----------------------------------------------------------------------------
__global__ __launch_bounds__(256)
void cca_proj_kernel(const float* __restrict__ x1, const float* __restrict__ x2,
                     const float* __restrict__ wq, const float* __restrict__ bq,
                     const float* __restrict__ wk, const float* __restrict__ bk,
                     const float* __restrict__ wv, const float* __restrict__ bv,
                     _Float16* __restrict__ Qh, _Float16* __restrict__ Kh,
                     _Float16* __restrict__ Vt)
{
    // two input buffers of (A:128x40 + B:128x40) halfs = 10240 halfs each;
    // epilogue reuses the same space as a 128x136 output tile (17408 halfs).
    __shared__ alignas(16) _Float16 smem[20480];

    const int tid  = threadIdx.x;
    const int lane = tid & 31;
    const int wave = tid >> 5;
    const int lm   = lane & 15;
    const int h    = lane >> 4;

    const int z = blockIdx.z;
    const int b = z / 3;
    const int s = z - 3 * b;

    const float* X    = (s == 0 ? x1 : x2) + (size_t)b * CH * NSP;
    const float* W    = (s == 0 ? wq : (s == 1 ? wk : wv));
    const float* bias = (s == 0 ? bq : (s == 1 ? bk : bv));

    const int mBlock = blockIdx.y * 128;
    const int nBlock = blockIdx.x * 128;
    const int mW = (wave >> 1) * 32;   // wave M offset in block tile
    const int nW = (wave & 1) * 64;    // wave N offset in block tile

    v8f acc[2][4];
    #pragma unroll
    for (int i = 0; i < 2; ++i)
        #pragma unroll
        for (int j = 0; j < 4; ++j) acc[i][j] = (v8f){0,0,0,0,0,0,0,0};

    // loader mappings
    const int wr = tid >> 1;          // W tile row 0..127
    const int wh = tid & 1;           // W tile k-half (16 floats)
    const int xk = tid >> 3;          // X tile k row 0..31
    const int xg = tid & 7;           // X tile n group (16 floats)

    const float* wbase = W + (size_t)(mBlock + wr) * CH + wh * 16;
    const float* xbase = X + (size_t)xk * NSP + nBlock + xg * 16;

    float4 w0, w1, w2, w3, xv0, xv1, xv2, xv3;

    // prologue: load k0 = 0 and stage into buffer 0
    w0  = *(const float4*)(wbase + 0);
    w1  = *(const float4*)(wbase + 4);
    w2  = *(const float4*)(wbase + 8);
    w3  = *(const float4*)(wbase + 12);
    xv0 = *(const float4*)(xbase + 0);
    xv1 = *(const float4*)(xbase + 4);
    xv2 = *(const float4*)(xbase + 8);
    xv3 = *(const float4*)(xbase + 12);
    {
        _Float16* dst = smem;            // buffer 0
        HPack8 p0, p1;
        p0.h[0]=(_Float16)w0.x; p0.h[1]=(_Float16)w0.y; p0.h[2]=(_Float16)w0.z; p0.h[3]=(_Float16)w0.w;
        p0.h[4]=(_Float16)w1.x; p0.h[5]=(_Float16)w1.y; p0.h[6]=(_Float16)w1.z; p0.h[7]=(_Float16)w1.w;
        p1.h[0]=(_Float16)w2.x; p1.h[1]=(_Float16)w2.y; p1.h[2]=(_Float16)w2.z; p1.h[3]=(_Float16)w2.w;
        p1.h[4]=(_Float16)w3.x; p1.h[5]=(_Float16)w3.y; p1.h[6]=(_Float16)w3.z; p1.h[7]=(_Float16)w3.w;
        *(uint4*)&dst[wr * 40 + wh * 16 + 0] = p0.u;
        *(uint4*)&dst[wr * 40 + wh * 16 + 8] = p1.u;
        _Float16* bd = dst + 5120;
        const int nb = xg * 16;
        bd[(nb+ 0)*40+xk]=(_Float16)xv0.x; bd[(nb+ 1)*40+xk]=(_Float16)xv0.y;
        bd[(nb+ 2)*40+xk]=(_Float16)xv0.z; bd[(nb+ 3)*40+xk]=(_Float16)xv0.w;
        bd[(nb+ 4)*40+xk]=(_Float16)xv1.x; bd[(nb+ 5)*40+xk]=(_Float16)xv1.y;
        bd[(nb+ 6)*40+xk]=(_Float16)xv1.z; bd[(nb+ 7)*40+xk]=(_Float16)xv1.w;
        bd[(nb+ 8)*40+xk]=(_Float16)xv2.x; bd[(nb+ 9)*40+xk]=(_Float16)xv2.y;
        bd[(nb+10)*40+xk]=(_Float16)xv2.z; bd[(nb+11)*40+xk]=(_Float16)xv2.w;
        bd[(nb+12)*40+xk]=(_Float16)xv3.x; bd[(nb+13)*40+xk]=(_Float16)xv3.y;
        bd[(nb+14)*40+xk]=(_Float16)xv3.z; bd[(nb+15)*40+xk]=(_Float16)xv3.w;
    }
    __syncthreads();

    for (int it = 0; it < 16; ++it) {
        const int cur = it & 1;
        // issue next tile's global loads first (overlap with LDS/WMMA below)
        if (it < 15) {
            const int k1 = (it + 1) * 32;
            const float* wp = wbase + k1;
            w0 = *(const float4*)(wp + 0);
            w1 = *(const float4*)(wp + 4);
            w2 = *(const float4*)(wp + 8);
            w3 = *(const float4*)(wp + 12);
            const float* xp = xbase + (size_t)k1 * NSP;
            xv0 = *(const float4*)(xp + 0);
            xv1 = *(const float4*)(xp + 4);
            xv2 = *(const float4*)(xp + 8);
            xv3 = *(const float4*)(xp + 12);
        }

        // compute from buffer `cur`
        {
            const _Float16* ab = &smem[cur * 10240];
            const _Float16* bb = ab + 5120;
            HFrag aF[2], bF[4];
            #pragma unroll
            for (int tm = 0; tm < 2; ++tm) {
                const _Float16* p = ab + (mW + tm * 16 + lm) * 40 + h * 8;
                aF[tm].u[0] = *(const uint4*)(p);
                aF[tm].u[1] = *(const uint4*)(p + 16);
            }
            #pragma unroll
            for (int tn = 0; tn < 4; ++tn) {
                const _Float16* p = bb + (nW + tn * 16 + lm) * 40 + h * 8;
                bF[tn].u[0] = *(const uint4*)(p);
                bF[tn].u[1] = *(const uint4*)(p + 16);
            }
            #pragma unroll
            for (int tm = 0; tm < 2; ++tm)
                #pragma unroll
                for (int tn = 0; tn < 4; ++tn)
                    acc[tm][tn] = WMMA(aF[tm].v, bF[tn].v, acc[tm][tn]);
        }

        // stage next tile into the other buffer
        if (it < 15) {
            _Float16* dst = &smem[((it + 1) & 1) * 10240];
            HPack8 p0, p1;
            p0.h[0]=(_Float16)w0.x; p0.h[1]=(_Float16)w0.y; p0.h[2]=(_Float16)w0.z; p0.h[3]=(_Float16)w0.w;
            p0.h[4]=(_Float16)w1.x; p0.h[5]=(_Float16)w1.y; p0.h[6]=(_Float16)w1.z; p0.h[7]=(_Float16)w1.w;
            p1.h[0]=(_Float16)w2.x; p1.h[1]=(_Float16)w2.y; p1.h[2]=(_Float16)w2.z; p1.h[3]=(_Float16)w2.w;
            p1.h[4]=(_Float16)w3.x; p1.h[5]=(_Float16)w3.y; p1.h[6]=(_Float16)w3.z; p1.h[7]=(_Float16)w3.w;
            *(uint4*)&dst[wr * 40 + wh * 16 + 0] = p0.u;
            *(uint4*)&dst[wr * 40 + wh * 16 + 8] = p1.u;
            _Float16* bd = dst + 5120;
            const int nb = xg * 16;
            bd[(nb+ 0)*40+xk]=(_Float16)xv0.x; bd[(nb+ 1)*40+xk]=(_Float16)xv0.y;
            bd[(nb+ 2)*40+xk]=(_Float16)xv0.z; bd[(nb+ 3)*40+xk]=(_Float16)xv0.w;
            bd[(nb+ 4)*40+xk]=(_Float16)xv1.x; bd[(nb+ 5)*40+xk]=(_Float16)xv1.y;
            bd[(nb+ 6)*40+xk]=(_Float16)xv1.z; bd[(nb+ 7)*40+xk]=(_Float16)xv1.w;
            bd[(nb+ 8)*40+xk]=(_Float16)xv2.x; bd[(nb+ 9)*40+xk]=(_Float16)xv2.y;
            bd[(nb+10)*40+xk]=(_Float16)xv2.z; bd[(nb+11)*40+xk]=(_Float16)xv2.w;
            bd[(nb+12)*40+xk]=(_Float16)xv3.x; bd[(nb+13)*40+xk]=(_Float16)xv3.y;
            bd[(nb+14)*40+xk]=(_Float16)xv3.z; bd[(nb+15)*40+xk]=(_Float16)xv3.w;
        }
        __syncthreads();
    }

    // bias (depends on tm, h, r only)
    float bl[2][8];
    #pragma unroll
    for (int tm = 0; tm < 2; ++tm)
        #pragma unroll
        for (int r = 0; r < 8; ++r)
            bl[tm][r] = bias[mBlock + mW + tm * 16 + 8 * h + r];

    if (s < 2) {
        // transpose C tile through LDS -> coalesced b128 row stores
        #pragma unroll
        for (int tm = 0; tm < 2; ++tm)
            #pragma unroll
            for (int tn = 0; tn < 4; ++tn) {
                const int cn = nW + tn * 16 + lm;
                #pragma unroll
                for (int r = 0; r < 8; ++r)
                    smem[(mW + tm * 16 + 8 * h + r) * 136 + cn] =
                        (_Float16)(acc[tm][tn][r] + bl[tm][r]);
            }
        __syncthreads();
        _Float16* ob = (s == 0 ? Qh : Kh) + (size_t)b * CH * NSP;
        const int rr  = tid >> 1;
        const int seg = (tid & 1) * 64;
        _Float16* op = ob + (size_t)(mBlock + rr) * NSP + nBlock + seg;
        const _Float16* sp = &smem[rr * 136 + seg];
        #pragma unroll
        for (int q = 0; q < 8; ++q)
            *(uint4*)(op + q * 8) = *(const uint4*)(sp + q * 8);
    } else {
        _Float16* vb = Vt + (size_t)b * NSP * CH;
        #pragma unroll
        for (int tm = 0; tm < 2; ++tm)
            #pragma unroll
            for (int tn = 0; tn < 4; ++tn) {
                const int gn  = nBlock + nW + tn * 16 + lm;       // Vt row (n)
                const int gm0 = mBlock + mW + tm * 16 + 8 * h;    // 8 consecutive d
                HPack8 p;
                #pragma unroll
                for (int r = 0; r < 8; ++r)
                    p.h[r] = (_Float16)(acc[tm][tn][r] + bl[tm][r]);
                *(uint4*)&vb[(size_t)gn * CH + gm0] = p.u;
            }
    }
}

// -----------------------------------------------------------------------------
// K2: scores  S[b][c][d] = sum_n Q[b][c][n] * K[b][d][n]   (f16 in, f32 out)
// grid = (16 nUnits, 2 mGroups, 16 batch), block = 256.
// Each wave: 32x32 region = 2x2 WMMA tiles -> 2 global b128 loads per wmma.
// -----------------------------------------------------------------------------
__global__ __launch_bounds__(256)
void cca_scores_kernel(const _Float16* __restrict__ Qh,
                       const _Float16* __restrict__ Kh,
                       float* __restrict__ S)
{
    const int lane = threadIdx.x & 31;
    const int wave = threadIdx.x >> 5;
    const int lm = lane & 15, h = lane >> 4;
    const int b  = blockIdx.z;
    const int m0 = (blockIdx.y * 8 + wave) * 32;  // 0..480
    const int d0 = blockIdx.x * 32;               // 0..480

    const _Float16* pa0 = Qh + (size_t)b * CH * NSP + (size_t)(m0 + lm) * NSP + h * 8;
    const _Float16* pa1 = pa0 + (size_t)16 * NSP;
    const _Float16* pb0 = Kh + (size_t)b * CH * NSP + (size_t)(d0 + lm) * NSP + h * 8;
    const _Float16* pb1 = pb0 + (size_t)16 * NSP;

    v8f acc[2][2];
    #pragma unroll
    for (int i = 0; i < 2; ++i)
        #pragma unroll
        for (int j = 0; j < 2; ++j) acc[i][j] = (v8f){0,0,0,0,0,0,0,0};

    #pragma unroll 2
    for (int n0 = 0; n0 < NSP; n0 += 32) {
        if (n0 + 512 < NSP) {
            __builtin_prefetch(pa0 + n0 + 512, 0, 0);
            __builtin_prefetch(pa1 + n0 + 512, 0, 0);
            __builtin_prefetch(pb0 + n0 + 512, 0, 0);
            __builtin_prefetch(pb1 + n0 + 512, 0, 0);
        }
        HFrag a0, a1, b0, b1;
        a0.u[0] = *(const uint4*)(pa0 + n0);  a0.u[1] = *(const uint4*)(pa0 + n0 + 16);
        a1.u[0] = *(const uint4*)(pa1 + n0);  a1.u[1] = *(const uint4*)(pa1 + n0 + 16);
        b0.u[0] = *(const uint4*)(pb0 + n0);  b0.u[1] = *(const uint4*)(pb0 + n0 + 16);
        b1.u[0] = *(const uint4*)(pb1 + n0);  b1.u[1] = *(const uint4*)(pb1 + n0 + 16);
        acc[0][0] = WMMA(a0.v, b0.v, acc[0][0]);
        acc[0][1] = WMMA(a0.v, b1.v, acc[0][1]);
        acc[1][0] = WMMA(a1.v, b0.v, acc[1][0]);
        acc[1][1] = WMMA(a1.v, b1.v, acc[1][1]);
    }

    float* Sb = S + (size_t)b * CH * CH;
    #pragma unroll
    for (int tm = 0; tm < 2; ++tm)
        #pragma unroll
        for (int tn = 0; tn < 2; ++tn)
            #pragma unroll
            for (int r = 0; r < 8; ++r)
                Sb[(size_t)(m0 + tm * 16 + 8 * h + r) * CH + d0 + tn * 16 + lm] =
                    acc[tm][tn][r];
}

// -----------------------------------------------------------------------------
// K3: softmax over last dim (512) of S -> attn (f16). One wave per row.
// -----------------------------------------------------------------------------
__global__ __launch_bounds__(256)
void cca_softmax_kernel(const float* __restrict__ S, _Float16* __restrict__ attn)
{
    const int lane = threadIdx.x & 31;
    const int wave = threadIdx.x >> 5;
    const int row  = blockIdx.x * 8 + wave;   // 0 .. 16*512-1

    const float* rp = S + (size_t)row * CH + lane * 16;
    float4 v0 = *(const float4*)(rp + 0);
    float4 v1 = *(const float4*)(rp + 4);
    float4 v2 = *(const float4*)(rp + 8);
    float4 v3 = *(const float4*)(rp + 12);
    float v[16] = { v0.x,v0.y,v0.z,v0.w, v1.x,v1.y,v1.z,v1.w,
                    v2.x,v2.y,v2.z,v2.w, v3.x,v3.y,v3.z,v3.w };

    float m = v[0];
    #pragma unroll
    for (int i = 1; i < 16; ++i) m = fmaxf(m, v[i]);
    #pragma unroll
    for (int off = 16; off >= 1; off >>= 1) m = fmaxf(m, __shfl_xor(m, off, 32));

    float sum = 0.f;
    #pragma unroll
    for (int i = 0; i < 16; ++i) { v[i] = __expf(v[i] - m); sum += v[i]; }
    #pragma unroll
    for (int off = 16; off >= 1; off >>= 1) sum += __shfl_xor(sum, off, 32);
    const float inv = 1.0f / sum;

    union { uint4 u[2]; _Float16 h[16]; } o;
    #pragma unroll
    for (int i = 0; i < 16; ++i) o.h[i] = (_Float16)(v[i] * inv);
    _Float16* op = attn + (size_t)row * CH + lane * 16;
    *(uint4*)(op)     = o.u[0];
    *(uint4*)(op + 8) = o.u[1];
}

// -----------------------------------------------------------------------------
// K4: out[b][c][n] = sum_d attn[b][c][d] * Vt[b][n][d]   (f16 in, f32 out)
// grid = (64 nUnits, 2 mGroups, 16 batch), block = 256.
// Each wave: 32x64 region = 2x4 WMMA tiles -> 1.5 loads per wmma.
// D-layout gives coalesced f32 row stores (lanes 0..15 = consecutive n).
// -----------------------------------------------------------------------------
__global__ __launch_bounds__(256)
void cca_out_kernel(const _Float16* __restrict__ attn,
                    const _Float16* __restrict__ Vt,
                    float* __restrict__ out)
{
    const int lane = threadIdx.x & 31;
    const int wave = threadIdx.x >> 5;
    const int lm = lane & 15, h = lane >> 4;
    const int b  = blockIdx.z;
    const int m0 = (blockIdx.y * 8 + wave) * 32;  // c tile base (0..480)
    const int n0 = blockIdx.x * 64;               // n tile base (0..4032)

    const _Float16* pa0 = attn + (size_t)b * CH * CH + (size_t)(m0 + lm) * CH + h * 8;
    const _Float16* pa1 = pa0 + (size_t)16 * CH;
    const _Float16* pb  = Vt + (size_t)b * NSP * CH + (size_t)(n0 + lm) * CH + h * 8;

    v8f acc[2][4];
    #pragma unroll
    for (int i = 0; i < 2; ++i)
        #pragma unroll
        for (int j = 0; j < 4; ++j) acc[i][j] = (v8f){0,0,0,0,0,0,0,0};

    #pragma unroll 2
    for (int d0 = 0; d0 < CH; d0 += 32) {
        if (d0 + 256 < CH) {
            __builtin_prefetch(pa0 + d0 + 256, 0, 0);
            __builtin_prefetch(pa1 + d0 + 256, 0, 0);
        }
        HFrag a0, a1, bf[4];
        a0.u[0] = *(const uint4*)(pa0 + d0);  a0.u[1] = *(const uint4*)(pa0 + d0 + 16);
        a1.u[0] = *(const uint4*)(pa1 + d0);  a1.u[1] = *(const uint4*)(pa1 + d0 + 16);
        #pragma unroll
        for (int tn = 0; tn < 4; ++tn) {
            const _Float16* p = pb + (size_t)(tn * 16) * CH + d0;
            bf[tn].u[0] = *(const uint4*)(p);
            bf[tn].u[1] = *(const uint4*)(p + 16);
        }
        #pragma unroll
        for (int tn = 0; tn < 4; ++tn) {
            acc[0][tn] = WMMA(a0.v, bf[tn].v, acc[0][tn]);
            acc[1][tn] = WMMA(a1.v, bf[tn].v, acc[1][tn]);
        }
    }

    float* ob = out + (size_t)b * CH * NSP;
    #pragma unroll
    for (int tm = 0; tm < 2; ++tm)
        #pragma unroll
        for (int tn = 0; tn < 4; ++tn)
            #pragma unroll
            for (int r = 0; r < 8; ++r)
                ob[(size_t)(m0 + tm * 16 + 8 * h + r) * NSP + n0 + tn * 16 + lm] =
                    acc[tm][tn][r];
}

// -----------------------------------------------------------------------------
extern "C" void kernel_launch(void* const* d_in, const int* in_sizes, int n_in,
                              void* d_out, int out_size, void* d_ws, size_t ws_size,
                              hipStream_t stream)
{
    const float* x1 = (const float*)d_in[0];
    const float* x2 = (const float*)d_in[1];
    const float* wq = (const float*)d_in[2];
    const float* bq = (const float*)d_in[3];
    const float* wk = (const float*)d_in[4];
    const float* bk = (const float*)d_in[5];
    const float* wv = (const float*)d_in[6];
    const float* bv = (const float*)d_in[7];
    float* out = (float*)d_out;

    const size_t QB = (size_t)BATCH * CH * NSP * sizeof(_Float16); // 64 MB
    const size_t SB = (size_t)BATCH * CH * CH * sizeof(float);     // 16 MB

    char* ws = (char*)d_ws;
    _Float16* Qh   = (_Float16*)(ws);
    _Float16* Kh   = (_Float16*)(ws + QB);
    _Float16* Vt   = (_Float16*)(ws + 2 * QB);
    float*    S    = (float*)   (ws + 3 * QB);
    _Float16* attn = (_Float16*)(ws + 3 * QB + SB);

    cca_proj_kernel<<<dim3(NSP / 128, CH / 128, BATCH * 3), 256, 0, stream>>>(
        x1, x2, wq, bq, wk, bk, wv, bv, Qh, Kh, Vt);
    cca_scores_kernel<<<dim3(CH / 32, 2, BATCH), 256, 0, stream>>>(Qh, Kh, S);
    cca_softmax_kernel<<<dim3(BATCH * CH / 8), 256, 0, stream>>>(S, attn);
    cca_out_kernel<<<dim3(NSP / 64, 2, BATCH), 256, 0, stream>>>(attn, Vt, out);
}
